// Net_61847529062502
// MI455X (gfx1250) — matrix operand order repaired and checked by reading, not compiled, not run on hardware
//
#include <hip/hip_runtime.h>
#include <cstdint>
#include <cstddef>

// ---------------------------------------------------------------------------
// CDNA5 / gfx1250 implementation of the conv + 2-layer biLSTM + attention net.
// Matrix work (input projections + recurrent matvecs) runs on
// v_wmma_f32_16x16x32_bf16 (wave32, 16x16 tiles, f32 accumulation).
// B-tile staging uses the gfx1250 async Global->LDS path when available.
// ---------------------------------------------------------------------------

typedef __attribute__((ext_vector_type(16))) __bf16 v16bf;
typedef __attribute__((ext_vector_type(8)))  float  v8f;
typedef int v4i_vs __attribute__((vector_size(16)));   // b128 payload type

union FragBF { v16bf v; float4 q[2]; };

#define B_   32
#define T_   512
#define H_   512
#define G4_  2048      // 4*H, gate width
#define BT_  16384     // B*T

#if __has_builtin(__builtin_amdgcn_global_load_async_to_lds_b128) && \
    __has_builtin(__builtin_amdgcn_s_wait_asynccnt)
#define USE_ASYNC_LDS 1
#else
#define USE_ASYNC_LDS 0
#endif

__device__ inline __bf16 f2bf(float f) {
  unsigned u = __float_as_uint(f);
  u = (u + 0x7FFFu + ((u >> 16) & 1u)) >> 16;   // round-to-nearest-even
  unsigned short s = (unsigned short)u;
  __bf16 r;
  __builtin_memcpy(&r, &s, 2);
  return r;
}

__device__ inline float sigf(float x) { return 1.0f / (1.0f + __expf(-x)); }

// --------------------------- f32 -> bf16 convert ---------------------------
__global__ void k_cvt_bf16(const float* __restrict__ s, __bf16* __restrict__ d, int n) {
  int i = blockIdx.x * blockDim.x + threadIdx.x;
  if (i < n) d[i] = f2bf(s[i]);
}

__global__ void k_zero_f32(float* __restrict__ p, int n) {
  int i = blockIdx.x * blockDim.x + threadIdx.x;
  if (i < n) p[i] = 0.0f;
}

// ------------------- conv1d(3->6,k3,p1)+BN+ReLU+maxpool2 -------------------
// Produces layer-0 LSTM input rows: [b*T+t] x 256 (192 conv feat | 64 x2), bf16.
__global__ void k_conv_feat(const float* __restrict__ x1, const float* __restrict__ x2,
                            const float* __restrict__ cw, const float* __restrict__ cb,
                            const float* __restrict__ gam, const float* __restrict__ bet,
                            const float* __restrict__ mu,  const float* __restrict__ var,
                            __bf16* __restrict__ Xin) {
  int idx = blockIdx.x * blockDim.x + threadIdx.x;      // BT*256 threads
  if (idx >= BT_ * 256) return;
  int col = idx & 255;
  int bt  = idx >> 8;
  float v;
  if (col < 192) {
    int ch = col >> 5;          // 0..5
    int p  = col & 31;          // pooled position
    const float* xin = x1 + (size_t)bt * 3 * 64;
    float acc0 = cb[ch], acc1 = cb[ch];
    int l0 = 2 * p, l1 = 2 * p + 1;
#pragma unroll
    for (int c = 0; c < 3; ++c) {
      const float* xr = xin + c * 64;
      const float* wr = cw + (ch * 3 + c) * 3;
      float a0 = (l0 > 0) ? xr[l0 - 1] : 0.0f;
      acc0 += wr[0] * a0 + wr[1] * xr[l0] + wr[2] * xr[l0 + 1];
      float c1 = (l1 < 63) ? xr[l1 + 1] : 0.0f;
      acc1 += wr[0] * xr[l1 - 1] + wr[1] * xr[l1] + wr[2] * c1;
    }
    float scale = rsqrtf(var[ch] + 1e-5f) * gam[ch];
    acc0 = fmaxf((acc0 - mu[ch]) * scale + bet[ch], 0.0f);
    acc1 = fmaxf((acc1 - mu[ch]) * scale + bet[ch], 0.0f);
    v = fmaxf(acc0, acc1);
  } else {
    v = x2[(size_t)bt * 64 + (col - 192)];
  }
  Xin[idx] = f2bf(v);
}

// --------------------- big input-projection GEMM (WMMA) ---------------------
// G[m,n] = sum_k X[m,k]*W[n,k] + b_ih[n] + b_hh[n].  M=16384, N=2048, K param.
// Block = 256 thr = 8 waves; block tile = 128M x 64N; B tile staged into LDS
// via async Global->LDS copies (ASYNCcnt) when the toolchain exposes them.
__global__ void k_gemm_ih(const __bf16* __restrict__ X, int K,
                          const __bf16* __restrict__ Wf, const __bf16* __restrict__ Wb,
                          const float* __restrict__ bihf, const float* __restrict__ bhhf,
                          const float* __restrict__ bihb, const float* __restrict__ bhhb,
                          float* __restrict__ Gf, float* __restrict__ Gb) {
  const __bf16* W  = blockIdx.z ? Wb   : Wf;
  const float*  b1 = blockIdx.z ? bihb : bihf;
  const float*  b2 = blockIdx.z ? bhhb : bhhf;
  float*        G  = blockIdx.z ? Gb   : Gf;

  int wave = threadIdx.x >> 5;
  int lane = threadIdx.x & 31;            // wave32
  int m0 = blockIdx.x * 128 + wave * 16;
  int n0 = blockIdx.y * 64;

  __shared__ __align__(16) __bf16 Bs[64 * 32];   // 64 N-rows x 32 K, 4 KB

  v8f acc[4] = {};
  int ldrow = threadIdx.x >> 2;           // 0..63 : N row within tile
  int ldk   = (threadIdx.x & 3) * 8;      // K chunk (8 bf16 = 16 B)

  int am = lane & 15;                     // A: row within tile
  int ak = (lane >> 4) << 3;              // A: kbase 0/8
  int bk = (lane >> 4) << 3;              // B: kbase 0/8

  for (int k0 = 0; k0 < K; k0 += 32) {
    const __bf16* gsrc = &W[(size_t)(n0 + ldrow) * K + k0 + ldk];
    __bf16*       ldst = &Bs[ldrow * 32 + ldk];
#if USE_ASYNC_LDS
    // direct Global->LDS DMA, no VGPR round-trip; tracked by ASYNCcnt
    __builtin_amdgcn_global_load_async_to_lds_b128(
        (__attribute__((address_space(1))) v4i_vs*)gsrc,
        (__attribute__((address_space(3))) v4i_vs*)ldst,
        0, 0);
    if (k0 + 32 < K)
      __builtin_prefetch(gsrc + 32, 0, 1);
    __builtin_amdgcn_s_wait_asynccnt(0);   // this wave's copies landed in LDS
    __syncthreads();                        // ...and every other wave's too
#else
    *reinterpret_cast<float4*>(ldst) = *reinterpret_cast<const float4*>(gsrc);
    if (k0 + 32 < K)
      __builtin_prefetch(gsrc + 32, 0, 1);
    __syncthreads();
#endif

    FragBF a;
    {
      const __bf16* p = X + (size_t)(m0 + am) * K + k0 + ak;
      a.q[0] = *reinterpret_cast<const float4*>(p);        // K = kbase..kbase+7
      a.q[1] = *reinterpret_cast<const float4*>(p + 16);   // K = 16+kbase..
    }
#pragma unroll
    for (int nn = 0; nn < 4; ++nn) {
      FragBF bf;
      const __bf16* p = &Bs[(nn * 16 + am) * 32 + bk];
      bf.q[0] = *reinterpret_cast<const float4*>(p);       // K = kbase..kbase+7
      bf.q[1] = *reinterpret_cast<const float4*>(p + 16);  // K = 16+kbase..
      acc[nn] = __builtin_amdgcn_wmma_f32_16x16x32_bf16(
          false, a.v, false, bf.v, (short)0, acc[nn], false, false);
    }
    __syncthreads();
  }

  int mb = m0 + ((lane >> 4) << 3);       // D layout: VGPR j -> M = mb+j
#pragma unroll
  for (int nn = 0; nn < 4; ++nn) {
    int n = n0 + nn * 16 + am;
    float bias = b1[n] + b2[n];
#pragma unroll
    for (int j = 0; j < 8; ++j)
      G[(size_t)(mb + j) * G4_ + n] = acc[nn][j] + bias;
  }
}

// ------------------ per-timestep recurrent GEMM (WMMA) ---------------------
// Ghh[dir, 32, 2048] = h_bf16[dir] (32x512) @ Whh(dir)^T.  512 16x16 tiles.
__global__ void k_gemm_hh(const __bf16* __restrict__ hB,
                          const __bf16* __restrict__ Wf, const __bf16* __restrict__ Wb,
                          float* __restrict__ Ghh) {
  int wid  = blockIdx.x * (blockDim.x >> 5) + (threadIdx.x >> 5);  // 0..511
  int lane = threadIdx.x & 31;
  int dir = wid >> 8;
  int rem = wid & 255;
  int mt  = rem >> 7;            // 0..1 (M tiles of 16 over B=32)
  int nt  = rem & 127;           // 0..127 (N tiles over 2048)

  const __bf16* A = hB + (size_t)dir * B_ * H_;
  const __bf16* W = dir ? Wb : Wf;
  float*        G = Ghh + (size_t)dir * B_ * G4_;

  int am = lane & 15;
  int ak = (lane >> 4) << 3;
  int bk = (lane >> 4) << 3;
  int m0 = mt * 16, n0 = nt * 16;

  v8f acc = {};
  for (int k0 = 0; k0 < H_; k0 += 32) {
    FragBF a, b;
    const __bf16* pa = A + (size_t)(m0 + am) * H_ + k0 + ak;
    a.q[0] = *reinterpret_cast<const float4*>(pa);
    a.q[1] = *reinterpret_cast<const float4*>(pa + 16);
    const __bf16* pb = W + (size_t)(n0 + am) * H_ + k0 + bk;
    b.q[0] = *reinterpret_cast<const float4*>(pb);
    b.q[1] = *reinterpret_cast<const float4*>(pb + 16);
    acc = __builtin_amdgcn_wmma_f32_16x16x32_bf16(
        false, a.v, false, b.v, (short)0, acc, false, false);
  }
  int n  = n0 + am;
  int mb = m0 + ((lane >> 4) << 3);
#pragma unroll
  for (int j = 0; j < 8; ++j)
    G[(size_t)(mb + j) * G4_ + n] = acc[j];
}

// ------------------------- LSTM cell (both dirs) ---------------------------
// Backward direction maps inner step s -> original t = len-1-s (no reversal).
__global__ void k_cell(int s,
                       const float* __restrict__ GihF, const float* __restrict__ GihB,
                       const float* __restrict__ Ghh,  const int* __restrict__ len,
                       float* __restrict__ cS, __bf16* __restrict__ hBf,
                       __bf16* __restrict__ Ybf, float* __restrict__ Yf) {
  int idx = blockIdx.x * blockDim.x + threadIdx.x;   // 2*32*512
  if (idx >= 2 * B_ * H_) return;
  int j   = idx & (H_ - 1);
  int b   = (idx >> 9) & (B_ - 1);
  int dir = idx >> 14;
  int L = len[b];
  bool valid = s < L;
  int t = dir ? (valid ? (L - 1 - s) : 0) : s;

  const float* Gih = dir ? GihB : GihF;
  size_t grow = ((size_t)b * T_ + t) * G4_;
  const float* gh = Ghh + (size_t)dir * B_ * G4_ + (size_t)b * G4_;

  float gi = Gih[grow + j]            + gh[j];
  float gf = Gih[grow + H_ + j]       + gh[H_ + j];
  float gg = Gih[grow + 2 * H_ + j]   + gh[2 * H_ + j];
  float go = Gih[grow + 3 * H_ + j]   + gh[3 * H_ + j];

  int hidx = dir * B_ * H_ + b * H_ + j;
  float c  = cS[hidx];
  float cn = sigf(gf) * c + sigf(gi) * tanhf(gg);
  float hn = sigf(go) * tanhf(cn);

  if (valid) {
    cS[hidx]  = cn;
    hBf[hidx] = f2bf(hn);
    size_t yi = ((size_t)b * T_ + t) * 1024 + (size_t)dir * H_ + j;
    if (Ybf) Ybf[yi] = f2bf(hn);
    if (Yf)  Yf[yi]  = hn;
  }
}

// -------------------- masked softmax attention pooling ---------------------
__global__ void k_att_pool(const float* __restrict__ Yf, const float* __restrict__ aw,
                           const int* __restrict__ len, float* __restrict__ pooled) {
  int b = blockIdx.x, tid = threadIdx.x;             // 256 threads
  __shared__ float ps[T_];
  __shared__ float red[256];
  int L = len[b];
  const float* Yb = Yf + (size_t)b * T_ * 1024;

  for (int t = tid; t < T_; t += 256) {
    if (t < L) {
      const float* yr = Yb + (size_t)t * 1024;
      float d = 0.0f;
      for (int j = 0; j < 1024; ++j) d += yr[j] * aw[j];
      ps[t] = d;
    } else ps[t] = -INFINITY;
  }
  __syncthreads();

  float m = -INFINITY;
  for (int t = tid; t < T_; t += 256) m = fmaxf(m, ps[t]);
  red[tid] = m; __syncthreads();
  for (int st = 128; st > 0; st >>= 1) {
    if (tid < st) red[tid] = fmaxf(red[tid], red[tid + st]);
    __syncthreads();
  }
  m = red[0]; __syncthreads();

  float sum = 0.0f;
  for (int t = tid; t < T_; t += 256) {
    float e = (ps[t] == -INFINITY) ? 0.0f : __expf(ps[t] - m);
    ps[t] = e; sum += e;
  }
  red[tid] = sum; __syncthreads();
  for (int st = 128; st > 0; st >>= 1) {
    if (tid < st) red[tid] += red[tid + st];
    __syncthreads();
  }
  float inv = 1.0f / red[0];
  __syncthreads();

  for (int j = tid; j < 1024; j += 256) {
    float acc = 0.0f;
    for (int t = 0; t < L; ++t) acc += ps[t] * Yb[(size_t)t * 1024 + j];
    pooled[b * 1024 + j] = acc * inv;
  }
}

// ------------------------ FC (64) + log_softmax ----------------------------
__global__ void k_fc_lsm(const float* __restrict__ pooled, const float* __restrict__ fw,
                         const float* __restrict__ fb, const int* __restrict__ len,
                         float* __restrict__ out, int out_size) {
  int b = blockIdx.x, o = threadIdx.x;               // 64 threads
  __shared__ float red[64];
  const float* p  = pooled + b * 1024;
  const float* wr = fw + o * 1024;
  float acc = fb[o];
  for (int j = 0; j < 1024; ++j) acc += p[j] * wr[j];
  red[o] = acc; __syncthreads();
  for (int st = 32; st > 0; st >>= 1) {
    if (o < st) red[o] = fmaxf(red[o], red[o + st]);
    __syncthreads();
  }
  float m = red[0]; __syncthreads();
  red[o] = __expf(acc - m); __syncthreads();
  for (int st = 32; st > 0; st >>= 1) {
    if (o < st) red[o] += red[o + st];
    __syncthreads();
  }
  out[b * 64 + o] = acc - m - __logf(red[0]);
  if (o == 0 && out_size >= B_ * 64 + B_)
    out[B_ * 64 + b] = __int_as_float(len[b]);       // second tuple output (bit-cast)
}

// ---------------------------------------------------------------------------
static void* carve(char*& p, size_t bytes) {
  void* r = p;
  p += (bytes + 255) & ~(size_t)255;
  return r;
}

extern "C" void kernel_launch(void* const* d_in, const int* in_sizes, int n_in,
                              void* d_out, int out_size, void* d_ws, size_t ws_size,
                              hipStream_t stream) {
  (void)in_sizes; (void)n_in; (void)ws_size;
  // setup_inputs dict order; lstm_params pytree-flattened to 16 leaves:
  // ((wih,whh,bih,bhh) fwd, bwd) x 2 layers.
  const float* x1      = (const float*)d_in[0];
  const float* x2      = (const float*)d_in[1];
  const int*   length  = (const int*)  d_in[2];
  const float* conv_w  = (const float*)d_in[3];
  const float* conv_b  = (const float*)d_in[4];
  const float* bn_g    = (const float*)d_in[5];
  const float* bn_b    = (const float*)d_in[6];
  const float* bn_m    = (const float*)d_in[7];
  const float* bn_v    = (const float*)d_in[8];
  const float *Wih[2][2], *Whh[2][2], *Bih[2][2], *Bhh[2][2];
  int pi = 9;
  for (int l = 0; l < 2; ++l)
    for (int d = 0; d < 2; ++d) {
      Wih[l][d] = (const float*)d_in[pi++];
      Whh[l][d] = (const float*)d_in[pi++];
      Bih[l][d] = (const float*)d_in[pi++];
      Bhh[l][d] = (const float*)d_in[pi++];
    }
  const float* att_w = (const float*)d_in[25];
  const float* fc_w  = (const float*)d_in[26];
  const float* fc_b  = (const float*)d_in[27];

  // ---- workspace carve ----
  char* wp = (char*)d_ws;
  __bf16* Xin  = (__bf16*)carve(wp, (size_t)BT_ * 256 * 2);
  __bf16* Y01  = (__bf16*)carve(wp, (size_t)BT_ * 1024 * 2);
  float*  Yf   = (float*) carve(wp, (size_t)BT_ * 1024 * 4);
  float*  GihF = (float*) carve(wp, (size_t)BT_ * G4_ * 4);
  float*  GihB = (float*) carve(wp, (size_t)BT_ * G4_ * 4);
  float*  Ghh  = (float*) carve(wp, (size_t)2 * B_ * G4_ * 4);
  float*  cS   = (float*) carve(wp, (size_t)2 * B_ * H_ * 4);
  __bf16* hBf  = (__bf16*)carve(wp, (size_t)2 * B_ * H_ * 2);
  float*  pooled = (float*)carve(wp, (size_t)B_ * 1024 * 4);
  const int ind[2] = {256, 1024};
  __bf16 *wihB[2][2], *whhB[2][2];
  for (int l = 0; l < 2; ++l)
    for (int d = 0; d < 2; ++d) {
      wihB[l][d] = (__bf16*)carve(wp, (size_t)G4_ * ind[l] * 2);
      whhB[l][d] = (__bf16*)carve(wp, (size_t)G4_ * H_ * 2);
    }

  // ---- weight conversion f32 -> bf16 ----
  for (int l = 0; l < 2; ++l)
    for (int d = 0; d < 2; ++d) {
      int n1 = G4_ * ind[l];
      k_cvt_bf16<<<(n1 + 255) / 256, 256, 0, stream>>>(Wih[l][d], wihB[l][d], n1);
      int n2 = G4_ * H_;
      k_cvt_bf16<<<(n2 + 255) / 256, 256, 0, stream>>>(Whh[l][d], whhB[l][d], n2);
    }

  // ---- conv front-end -> layer-0 input ----
  k_conv_feat<<<(BT_ * 256 + 255) / 256, 256, 0, stream>>>(
      x1, x2, conv_w, conv_b, bn_g, bn_b, bn_m, bn_v, Xin);

  // ---- two stacked bidirectional LSTM layers ----
  for (int layer = 0; layer < 2; ++layer) {
    int K = ind[layer];
    const __bf16* Xl = layer ? Y01 : Xin;
    k_zero_f32<<<(2 * B_ * H_ + 255) / 256, 256, 0, stream>>>(cS, 2 * B_ * H_);
    k_zero_f32<<<(2 * B_ * H_ / 2 + 255) / 256, 256, 0, stream>>>((float*)hBf, 2 * B_ * H_ / 2);

    dim3 g1(BT_ / 128, G4_ / 64, 2);
    k_gemm_ih<<<g1, 256, 0, stream>>>(Xl, K,
        wihB[layer][0], wihB[layer][1],
        Bih[layer][0], Bhh[layer][0], Bih[layer][1], Bhh[layer][1],
        GihF, GihB);

    for (int s = 0; s < T_; ++s) {
      k_gemm_hh<<<64, 256, 0, stream>>>(hBf, whhB[layer][0], whhB[layer][1], Ghh);
      k_cell<<<(2 * B_ * H_) / 256, 256, 0, stream>>>(
          s, GihF, GihB, Ghh, length, cS, hBf,
          layer == 0 ? Y01 : (__bf16*)nullptr,
          layer == 1 ? Yf  : (float*)nullptr);
    }
  }

  // ---- attention pooling + FC + log_softmax ----
  k_att_pool<<<B_, 256, 0, stream>>>(Yf, att_w, length, pooled);
  k_fc_lsm<<<B_, 64, 0, stream>>>(pooled, fc_w, fc_b, length, (float*)d_out, out_size);
}
